// xLSTM_8985071583369
// MI455X (gfx1250) — matrix operand order repaired and checked by reading, not compiled, not run on hardware
//
#include <hip/hip_runtime.h>

#define T_ 128
#define B_ 32
#define D_ 1024
#define H_ 1024
#define NH_ 8
#define PFS_T 1365   // true sLSTM MLP width
#define PFS_P 1376   // row-storage padding (mult of 32, for K of down-proj)
#define PFS_N 1536   // packed-N padding (mult of 256 cols -> NT4 branch-free)

typedef __attribute__((ext_vector_type(16))) __bf16 v16bf;
typedef __attribute__((ext_vector_type(8)))  float  v8f;

// ---------------------------------------------------------------- utilities
__device__ __forceinline__ unsigned short f2bf(float f) {
  unsigned int u = __float_as_uint(f);
  u += 0x7FFFu + ((u >> 16) & 1u);         // round-to-nearest-even
  return (unsigned short)(u >> 16);
}
__device__ __forceinline__ float sigmf(float x) { return 1.f / (1.f + __expf(-x)); }

// ---------------------------------------------------------------- pack kernels
// fp32 row-major weight [K x N] (per head) -> 16-bit B fragments (this file's contiguous k order):
// frag(nt,kt) at ((nt*nkt+kt)*512); lane = n%16 + 16*((k%32)/16); e = k%16.
__global__ void k_pack_w(const float* __restrict__ src, unsigned short* __restrict__ dst,
                         int K, int N, int Kp, int Np, long sHS, long dHS) {
  long idx = (long)blockIdx.x * 256 + threadIdx.x;
  long tot = (long)Kp * Np;
  if (idx >= tot) return;
  src += (long)blockIdx.z * sHS;
  dst += (long)blockIdx.z * dHS;
  int k = (int)(idx / Np), n = (int)(idx % Np);
  float v = (k < K && n < N) ? src[(long)k * N + n] : 0.f;
  int nkt = Kp >> 5;
  int nt = n >> 4, nl = n & 15, kt = k >> 5, kl = k & 31;
  int lane = nl + ((kl >> 4) << 4);
  int e = kl & 15;
  dst[((long)(nt * nkt + kt) << 9) + (lane << 4) + e] = f2bf(v);
}

// fp32 activation [32 x K] -> 16-bit A-fragment layout, kt-major:
// frag(kt,mt) at ((kt*2+mt)*512); lane = m%16 + 16*((k%32)/16); e = k%16.
__global__ void k_pack_a(const float* __restrict__ src, unsigned short* __restrict__ dst,
                         int K, int Kp) {
  int idx = blockIdx.x * 256 + threadIdx.x;
  int tot = 32 * Kp;
  if (idx >= tot) return;
  int m = idx / Kp, k = idx % Kp;
  float v = (k < K) ? src[m * K + k] : 0.f;
  int mt = m >> 4, ml = m & 15, kt = k >> 5, kl = k & 31;
  int g = kl >> 4;
  int lane = ml + (g << 4);
  int e = kl & 15;
  dst[((kt * 2 + mt) << 9) + (lane << 4) + e] = f2bf(v);
}

// ---------------------------------------------------------------- WMMA GEMM
// C[32 x *] = A[32 x K] @ W[K x N] + bias (+ res).  4 waves/block; each wave
// owns a 32(M) x (NT*16)(N) tile: 2*NT v_wmma_f32_16x16x32_bf16 per K-step,
// with the two A fragments reused across all NT weight fragments.
// NT=4 for dense GEMMs (ntTiles % 16 == 0 by construction -> branch-free),
// NT=2 for block-diagonal heads (8 tiles == one block).  grid.z = heads.
template <int NT>
__global__ __launch_bounds__(128) void k_gemm(
    const unsigned short* __restrict__ A, const unsigned short* __restrict__ W,
    const float* __restrict__ bias, const float* res, float* C,
    int nkt, int ntTiles, int Ntrue, int ldc,
    long aHS, long wHS, int bHS, int cHeadCol) {
  int lane = threadIdx.x & 31;
  int wv = threadIdx.x >> 5;
  int head = blockIdx.z;
  int nt0 = (blockIdx.x * 4 + wv) * NT;
  if (nt0 >= ntTiles) return;        // never taken for our launch geometry
  const v16bf* Af = (const v16bf*)(A + (long)head * aHS);
  const v16bf* Wb = (const v16bf*)(W + (long)head * wHS);
  const v16bf* Wp[NT];
#pragma unroll
  for (int i = 0; i < NT; ++i) Wp[i] = Wb + (long)(nt0 + i) * nkt * 32;
  v8f acc0[NT], acc1[NT];
#pragma unroll
  for (int i = 0; i < NT; ++i) { acc0[i] = (v8f){}; acc1[i] = (v8f){}; }
  for (int kt = 0; kt < nkt; ++kt) {
    if (kt + 4 < nkt) {                              // weight-stream prefetch
#pragma unroll
      for (int i = 0; i < NT; ++i)
        __builtin_prefetch((const void*)(Wp[i] + (kt + 4) * 32 + lane), 0, 1);
    }
    v16bf a0 = Af[kt * 64 + lane];
    v16bf a1 = Af[kt * 64 + 32 + lane];
#pragma unroll
    for (int i = 0; i < NT; ++i) {
      v16bf w = Wp[i][kt * 32 + lane];
      acc0[i] = __builtin_amdgcn_wmma_f32_16x16x32_bf16(false, a0, false, w, (short)0, acc0[i], false, false);
      acc1[i] = __builtin_amdgcn_wmma_f32_16x16x32_bf16(false, a1, false, w, (short)0, acc1[i], false, false);
    }
  }
  int nl = lane & 15, mh = (lane >> 4) * 8;
#pragma unroll
  for (int i = 0; i < NT; ++i) {
    int colh = (nt0 + i) * 16 + nl;
    if (colh >= Ntrue) continue;
    int col = head * cHeadCol + colh;
    float bv = bias ? bias[(long)head * bHS + colh] : 0.f;
#pragma unroll
    for (int r = 0; r < 8; ++r) {
      int m0 = r + mh;
      long i0 = (long)m0 * ldc + col;
      long i1 = (long)(m0 + 16) * ldc + col;
      float v0 = acc0[i][r] + bv;
      float v1 = acc1[i][r] + bv;
      if (res) { v0 += res[i0]; v1 += res[i1]; }
      C[i0] = v0; C[i1] = v1;
    }
  }
}

// ---------------------------------------------------------------- elementwise
__global__ __launch_bounds__(256) void k_ln(const float* __restrict__ x,
    const float* __restrict__ g, const float* __restrict__ b, float* __restrict__ y) {
  __shared__ float sA[256], sB[256];
  int row = blockIdx.x, tid = threadIdx.x;
  const float* xr = x + row * 1024;
  float s = 0.f, ss = 0.f;
  for (int j = tid; j < 1024; j += 256) { float v = xr[j]; s += v; ss += v * v; }
  sA[tid] = s; sB[tid] = ss; __syncthreads();
  for (int k = 128; k > 0; k >>= 1) {
    if (tid < k) { sA[tid] += sA[tid + k]; sB[tid] += sB[tid + k]; }
    __syncthreads();
  }
  float mu = sA[0] * (1.f / 1024.f);
  float var = sB[0] * (1.f / 1024.f) - mu * mu;
  float inv = rsqrtf(var + 1e-5f);
  for (int j = tid; j < 1024; j += 256)
    y[row * 1024 + j] = (xr[j] - mu) * inv * g[j] + b[j];
}

// causal conv over feature axis (taps 4) + SiLU
__global__ void k_conv_silu(const float* __restrict__ x, const float* __restrict__ w,
                            const float* __restrict__ cb, float* __restrict__ y,
                            int L, int total) {
  int idx = blockIdx.x * 256 + threadIdx.x;
  if (idx >= total) return;
  int b = idx / L, l = idx % L;
  const float* xr = x + b * L;
  float acc = cb[0] + w[3] * xr[l];
  if (l >= 1) acc += w[2] * xr[l - 1];
  if (l >= 2) acc += w[1] * xr[l - 2];
  if (l >= 3) acc += w[0] * xr[l - 3];
  y[idx] = acc * sigmf(acc);
}

// mLSTM exp-stabilized state update (k pre-scaled by 1/sqrt(HEAD))
__global__ void k_mgate(const float* __restrict__ it, const float* __restrict__ ft,
                        const float* __restrict__ kk, const float* __restrict__ vv,
                        float* cs, float* ns, float* ms) {
  int idx = blockIdx.x * 256 + threadIdx.x;
  if (idx >= 32 * 1024) return;
  float itv = it[idx], ftv = ft[idx], mp = ms[idx];
  float mt = fmaxf(ftv + mp, itv);
  float iv = __expf(itv - mt);
  float fv = __expf(ftv + mp - mt);
  float ks = kk[idx] * 0.0883883476483184f;   // 1/sqrt(128)
  cs[idx] = fv * cs[idx] + iv * (vv[idx] * ks);
  ns[idx] = fv * ns[idx] + iv * ks;
  ms[idx] = mt;
}

// denom[j] = max_l | sum_b n[b,j] * q[b,l] |
__global__ __launch_bounds__(128) void k_denom(const float* __restrict__ nst,
                                               const float* __restrict__ q,
                                               float* __restrict__ den) {
  __shared__ float nsh[32];
  __shared__ float red[128];
  int j = blockIdx.x, tid = threadIdx.x;
  if (tid < 32) nsh[tid] = nst[tid * 1024 + j];
  __syncthreads();
  float best = 0.f;
  for (int l = tid; l < 1024; l += 128) {
    float s = 0.f;
#pragma unroll
    for (int b = 0; b < 32; ++b) s += nsh[b] * q[b * 1024 + l];
    best = fmaxf(best, fabsf(s));
  }
  red[tid] = best; __syncthreads();
  for (int k = 64; k > 0; k >>= 1) {
    if (tid < k) red[tid] = fmaxf(red[tid], red[tid + k]);
    __syncthreads();
  }
  if (tid == 0) den[j] = red[0];
}

// mLSTM output: h = sig(og)*c*q/denom ; GroupNorm(8 groups) ; +skip ; *silu(xr)
__global__ __launch_bounds__(128) void k_mout(const float* __restrict__ og,
    const float* __restrict__ cs, const float* __restrict__ q,
    const float* __restrict__ den, const float* __restrict__ xs,
    const float* __restrict__ xr, const float* __restrict__ gg,
    const float* __restrict__ gb, float* __restrict__ out) {
  __shared__ float s1[128], s2[128];
  int b = blockIdx.x >> 3, grp = blockIdx.x & 7;
  int ch = threadIdx.x, j = grp * 128 + ch, idx = b * 1024 + j;
  float o = sigmf(og[idx]);
  float h = o * cs[idx] * q[idx] / den[j];
  s1[ch] = h; s2[ch] = h * h; __syncthreads();
  for (int k = 64; k > 0; k >>= 1) {
    if (ch < k) { s1[ch] += s1[ch + k]; s2[ch] += s2[ch + k]; }
    __syncthreads();
  }
  float mu = s1[0] * (1.f / 128.f);
  float var = s2[0] * (1.f / 128.f) - mu * mu;
  float gn = (h - mu) * rsqrtf(var + 1e-5f) * gg[j] + gb[j];
  float o1 = gn + xs[idx];
  float xv = xr[idx];
  out[idx] = o1 * (xv * sigmf(xv));
}

// sLSTM gate + state update + GroupNorm
__global__ __launch_bounds__(128) void k_sgate(const float* __restrict__ zr,
    const float* __restrict__ ir, const float* __restrict__ fr,
    const float* __restrict__ orw, float* cs, float* ns, float* ms, float* hs,
    const float* __restrict__ gg, const float* __restrict__ gb,
    float* __restrict__ gno) {
  __shared__ float s1[128], s2[128];
  int b = blockIdx.x >> 3, grp = blockIdx.x & 7;
  int ch = threadIdx.x, j = grp * 128 + ch, idx = b * 1024 + j;
  float z = tanhf(zr[idx]);
  float o = sigmf(orw[idx]);
  float itv = ir[idx], ftv = fr[idx], mp = ms[idx];
  float mt = fmaxf(ftv + mp, itv);
  float iv = __expf(itv - mt);
  float fv = __expf(ftv + mp - mt);
  float c = fv * cs[idx] + iv * z;
  float n = fv * ns[idx] + iv;
  float h = o * c / n;
  cs[idx] = c; ns[idx] = n; ms[idx] = mt; hs[idx] = h;
  s1[ch] = h; s2[ch] = h * h; __syncthreads();
  for (int k = 64; k > 0; k >>= 1) {
    if (ch < k) { s1[ch] += s1[ch + k]; s2[ch] += s2[ch + k]; }
    __syncthreads();
  }
  float mu = s1[0] * (1.f / 128.f);
  float var = s2[0] * (1.f / 128.f) - mu * mu;
  gno[idx] = (h - mu) * rsqrtf(var + 1e-5f) * gg[j] + gb[j];
}

__global__ void k_gelu_mul(const float* __restrict__ uL, const float* __restrict__ uR,
                           float* __restrict__ out, int total, int Np, int Ntrue) {
  int idx = blockIdx.x * 256 + threadIdx.x;
  if (idx >= total) return;
  int col = idx % Np;
  float r = 0.f;
  if (col < Ntrue) {
    float a = uL[idx], g = uR[idx];
    r = a * 0.5f * g * (1.f + erff(g * 0.7071067811865475f));
  }
  out[idx] = r;
}

__global__ __launch_bounds__(256) void k_proj(const float* __restrict__ X,
    const float* __restrict__ w, const float* __restrict__ b,
    float* __restrict__ out, int t) {
  __shared__ float s[256];
  int tid = threadIdx.x;
  float acc = 0.f;
  for (int j = tid; j < 1024; j += 256) acc += X[31 * 1024 + j] * w[j];
  s[tid] = acc; __syncthreads();
  for (int k = 128; k > 0; k >>= 1) {
    if (tid < k) s[tid] += s[tid + k];
    __syncthreads();
  }
  if (tid == 0) out[t] = s[0] + b[0];
}

__global__ void k_copy(const float* __restrict__ s, float* __restrict__ d, int n) {
  int i = blockIdx.x * 256 + threadIdx.x; if (i < n) d[i] = s[i];
}
__global__ void k_zero_f32(float* p, int n) {
  int i = blockIdx.x * 256 + threadIdx.x; if (i < n) p[i] = 0.f;
}
__global__ void k_zero_u16(unsigned short* p, int n) {
  int i = blockIdx.x * 256 + threadIdx.x; if (i < n) p[i] = 0;
}

// ---------------------------------------------------------------- host side
struct MParams {
  const float *Wf_b,*Wf_w,*Wi_b,*Wi_w,*Wk_b,*Wk_w,*Wo_b,*Wo_w,*Wq_b,*Wq_w,*Wv_b,*Wv_w;
  const float *conv_b,*conv_w,*down_b,*down_w,*gn_b,*gn_g,*ln_b,*ln_g,*skip_b,*skip_w;
  const float *upL_b,*upL_w,*upR_b,*upR_w;
};
struct SParams {
  const float *Rf_b,*Rf_w,*Ri_b,*Ri_w,*Ro_b,*Ro_w,*Rz_b,*Rz_w;
  const float *Wf_b,*Wf_w,*Wi_b,*Wi_w,*Wo_b,*Wo_w,*Wz_b,*Wz_w;
  const float *conv_b,*conv_w,*down_b,*down_w,*gn_b,*gn_g,*ln_b,*ln_g;
  const float *upL_b,*upL_w,*upR_b,*upR_w;
};
struct MPack { unsigned short *upL,*upR,*skip,*Wq,*Wk,*Wv,*Wi,*Wf,*Wo,*down; };
struct SPack { unsigned short *Wz,*Wi,*Wf,*Wo,*Rz,*Ri,*Rf,*Ro,*upL,*upR,*down; };

extern "C" void kernel_launch(void* const* d_in, const int* in_sizes, int n_in,
                              void* d_out, int out_size, void* d_ws, size_t ws_size,
                              hipStream_t stream) {
  (void)out_size;
  if (n_in < 111) return;

  // ---- locate x (unique size) and walk the param leaves (sorted-key order)
  int base, xi;
  if (in_sizes[0] == T_ * B_ * D_) { xi = 0; base = 1; }
  else { base = 0; xi = n_in - 1; }
  const float* x_in = (const float*)d_in[xi];
  int li = base;
  auto nx = [&]() -> const float* { return (const float*)d_in[li++]; };

  MParams mp[2]; SParams sp[2];
  for (int l = 0; l < 4; ++l) {
    if ((l & 1) == 0) {
      MParams& P = mp[l / 2];
      P.Wf_b=nx(); P.Wf_w=nx(); P.Wi_b=nx(); P.Wi_w=nx(); P.Wk_b=nx(); P.Wk_w=nx();
      P.Wo_b=nx(); P.Wo_w=nx(); P.Wq_b=nx(); P.Wq_w=nx(); P.Wv_b=nx(); P.Wv_w=nx();
      P.conv_b=nx(); P.conv_w=nx(); P.down_b=nx(); P.down_w=nx();
      P.gn_b=nx(); P.gn_g=nx(); P.ln_b=nx(); P.ln_g=nx();
      P.skip_b=nx(); P.skip_w=nx(); P.upL_b=nx(); P.upL_w=nx(); P.upR_b=nx(); P.upR_w=nx();
    } else {
      SParams& P = sp[l / 2];
      P.Rf_b=nx(); P.Rf_w=nx(); P.Ri_b=nx(); P.Ri_w=nx(); P.Ro_b=nx(); P.Ro_w=nx();
      P.Rz_b=nx(); P.Rz_w=nx();
      P.Wf_b=nx(); P.Wf_w=nx(); P.Wi_b=nx(); P.Wi_w=nx(); P.Wo_b=nx(); P.Wo_w=nx();
      P.Wz_b=nx(); P.Wz_w=nx();
      P.conv_b=nx(); P.conv_w=nx(); P.down_b=nx(); P.down_w=nx();
      P.gn_b=nx(); P.gn_g=nx(); P.ln_b=nx(); P.ln_g=nx();
      P.upL_b=nx(); P.upL_w=nx(); P.upR_b=nx(); P.upR_w=nx();
    }
  }
  const float* out_b = nx();
  const float* out_w = nx();

  // ---- workspace layout
  char* wsb = (char*)d_ws;
  size_t off = 0;
  auto take  = [&](size_t bytes) -> void* {
    size_t a = (off + 255) & ~(size_t)255; off = a + bytes; return (void*)(wsb + a);
  };
  auto takeF = [&](size_t n) -> float* { return (float*)take(n * 4); };
  auto takeU = [&](size_t n) -> unsigned short* { return (unsigned short*)take(n * 2); };

  MPack mk[2]; SPack sk[2];
  for (int i = 0; i < 2; ++i) {
    mk[i].upL  = takeU((size_t)1024 * 2048);
    mk[i].upR  = takeU((size_t)1024 * 1024);
    mk[i].skip = takeU((size_t)2048 * 1024);
    mk[i].Wq   = takeU((size_t)8 * 256 * 128);
    mk[i].Wk   = takeU((size_t)8 * 256 * 128);
    mk[i].Wv   = takeU((size_t)8 * 256 * 128);
    mk[i].Wi   = takeU((size_t)2048 * 1024);
    mk[i].Wf   = takeU((size_t)2048 * 1024);
    mk[i].Wo   = takeU((size_t)2048 * 1024);
    mk[i].down = takeU((size_t)1024 * 1024);
    sk[i].Wz = takeU((size_t)8 * 128 * 128); sk[i].Wi = takeU((size_t)8 * 128 * 128);
    sk[i].Wf = takeU((size_t)8 * 128 * 128); sk[i].Wo = takeU((size_t)8 * 128 * 128);
    sk[i].Rz = takeU((size_t)8 * 128 * 128); sk[i].Ri = takeU((size_t)8 * 128 * 128);
    sk[i].Rf = takeU((size_t)8 * 128 * 128); sk[i].Ro = takeU((size_t)8 * 128 * 128);
    sk[i].upL  = takeU((size_t)1024 * PFS_N);
    sk[i].upR  = takeU((size_t)1024 * PFS_N);
    sk[i].down = takeU((size_t)PFS_P * 1024);
  }

  float* X   = takeF(32768);
  float* xn  = takeF(32768);
  float* xl  = takeF(65536);
  float* xr  = takeF(32768);
  float* xc  = takeF(65536);
  float* xs  = takeF(32768);
  float* qb  = takeF(32768);
  float* kb  = takeF(32768);
  float* vb  = takeF(32768);
  float* t1  = takeF(32768);
  float* t2  = takeF(32768);
  float* t3  = takeF(32768);
  float* t4  = takeF(32768);
  float* dnm = takeF(1024);
  float* mo  = takeF(32768);
  float* gnb = takeF(32768);
  float* uL  = takeF((size_t)32 * PFS_P);
  float* uR  = takeF((size_t)32 * PFS_P);
  float* vp  = takeF((size_t)32 * PFS_P);
  float* states = takeF((size_t)4 * 4 * 32768);        // c,n,m,h per layer
  float *stC[4], *stN[4], *stM[4], *stH[4];
  for (int l = 0; l < 4; ++l) {
    stC[l] = states + (l * 4 + 0) * 32768; stN[l] = states + (l * 4 + 1) * 32768;
    stM[l] = states + (l * 4 + 2) * 32768; stH[l] = states + (l * 4 + 3) * 32768;
  }
  unsigned short* hpAll = takeU((size_t)4 * 32768);
  unsigned short* hp[4];
  for (int l = 0; l < 4; ++l) hp[l] = hpAll + (size_t)l * 32768;
  unsigned short* Xp  = takeU(32768);
  unsigned short* xnp = takeU(32768);
  unsigned short* xlp = takeU(65536);
  unsigned short* xcp = takeU(65536);
  unsigned short* mop = takeU(32768);
  unsigned short* gnp = takeU(32768);
  unsigned short* vpp = takeU((size_t)32 * PFS_P);
  if (off > ws_size) return;

  // ---- launch helpers
  auto packw = [&](const float* src, unsigned short* dst, int K, int N,
                   int Kp, int Np, int heads) {
    long tot = (long)Kp * Np;
    dim3 g((unsigned)((tot + 255) / 256), 1, (unsigned)heads);
    k_pack_w<<<g, 256, 0, stream>>>(src, dst, K, N, Kp, Np, (long)K * N, (long)Kp * Np);
  };
  auto packa = [&](const float* src, unsigned short* dst, int K, int Kp) {
    int tot = 32 * Kp;
    k_pack_a<<<dim3((tot + 255) / 256), 256, 0, stream>>>(src, dst, K, Kp);
  };
  auto gemm = [&](const unsigned short* A, const unsigned short* W, const float* bias,
                  const float* res, float* C, int Kp, int Np, int Ntrue, int ldc,
                  long aHS, long wHS, int bHS, int cHeadCol, int heads) {
    int nkt = Kp >> 5, ntT = Np >> 4;
    if ((ntT & 15) == 0) {           // dense path: 4 N-tiles per wave, no guards
      dim3 g((unsigned)(ntT / 16), 1, (unsigned)heads);
      k_gemm<4><<<g, 128, 0, stream>>>(A, W, bias, res, C, nkt, ntT, Ntrue, ldc,
                                       aHS, wHS, bHS, cHeadCol);
    } else {                         // block-diagonal heads: 8 tiles == 1 block
      dim3 g((unsigned)((ntT + 7) / 8), 1, (unsigned)heads);
      k_gemm<2><<<g, 128, 0, stream>>>(A, W, bias, res, C, nkt, ntT, Ntrue, ldc,
                                       aHS, wHS, bHS, cHeadCol);
    }
  };

  // ---- pack all weights to bf16 fragments (~75 MB -> resident in 192 MB L2)
  for (int i = 0; i < 2; ++i) {
    const MParams& P = mp[i]; const MPack& K = mk[i];
    packw(P.upL_w, K.upL, 1024, 2048, 1024, 2048, 1);
    packw(P.upR_w, K.upR, 1024, 1024, 1024, 1024, 1);
    packw(P.skip_w, K.skip, 2048, 1024, 2048, 1024, 1);
    packw(P.Wq_w, K.Wq, 256, 128, 256, 128, 8);
    packw(P.Wk_w, K.Wk, 256, 128, 256, 128, 8);
    packw(P.Wv_w, K.Wv, 256, 128, 256, 128, 8);
    packw(P.Wi_w, K.Wi, 2048, 1024, 2048, 1024, 1);
    packw(P.Wf_w, K.Wf, 2048, 1024, 2048, 1024, 1);
    packw(P.Wo_w, K.Wo, 2048, 1024, 2048, 1024, 1);
    packw(P.down_w, K.down, 1024, 1024, 1024, 1024, 1);
    const SParams& S = sp[i]; const SPack& Q = sk[i];
    packw(S.Wz_w, Q.Wz, 128, 128, 128, 128, 8);
    packw(S.Wi_w, Q.Wi, 128, 128, 128, 128, 8);
    packw(S.Wf_w, Q.Wf, 128, 128, 128, 128, 8);
    packw(S.Wo_w, Q.Wo, 128, 128, 128, 128, 8);
    packw(S.Rz_w, Q.Rz, 128, 128, 128, 128, 8);
    packw(S.Ri_w, Q.Ri, 128, 128, 128, 128, 8);
    packw(S.Rf_w, Q.Rf, 128, 128, 128, 128, 8);
    packw(S.Ro_w, Q.Ro, 128, 128, 128, 128, 8);
    packw(S.upL_w, Q.upL, 1024, PFS_T, 1024, PFS_N, 1);
    packw(S.upR_w, Q.upR, 1024, PFS_T, 1024, PFS_N, 1);
    packw(S.down_w, Q.down, PFS_T, 1024, PFS_P, 1024, 1);
  }

  // ---- zero recurrent state (deterministic per launch)
  {
    int nf = 4 * 4 * 32768;
    k_zero_f32<<<(nf + 255) / 256, 256, 0, stream>>>(states, nf);
    int nh = 4 * 32768;
    k_zero_u16<<<(nh + 255) / 256, 256, 0, stream>>>(hpAll, nh);
  }

  // ---- the sequential scan
  for (int t = 0; t < T_; ++t) {
    k_copy<<<128, 256, 0, stream>>>(x_in + (size_t)t * B_ * D_, X, 32768);
    for (int l = 0; l < 4; ++l) {
      if ((l & 1) == 0) {                      // ---------- mLSTM layer
        const MParams& P = mp[l / 2]; const MPack& K = mk[l / 2];
        k_ln<<<32, 256, 0, stream>>>(X, P.ln_g, P.ln_b, xn);
        packa(xn, xnp, 1024, 1024);
        gemm(xnp, K.upL, P.upL_b, nullptr, xl, 1024, 2048, 2048, 2048, 0, 0, 0, 0, 1);
        gemm(xnp, K.upR, P.upR_b, nullptr, xr, 1024, 1024, 1024, 1024, 0, 0, 0, 0, 1);
        k_conv_silu<<<(65536 + 255) / 256, 256, 0, stream>>>(xl, P.conv_w, P.conv_b, xc, 2048, 65536);
        packa(xc, xcp, 2048, 2048);
        packa(xl, xlp, 2048, 2048);
        gemm(xcp, K.skip, P.skip_b, nullptr, xs, 2048, 1024, 1024, 1024, 0, 0, 0, 0, 1);
        gemm(xcp, K.Wq, P.Wq_b, nullptr, qb, 256, 128, 128, 1024, 8192, 32768, 128, 128, 8);
        gemm(xcp, K.Wk, P.Wk_b, nullptr, kb, 256, 128, 128, 1024, 8192, 32768, 128, 128, 8);
        gemm(xlp, K.Wv, P.Wv_b, nullptr, vb, 256, 128, 128, 1024, 8192, 32768, 128, 128, 8);
        gemm(xcp, K.Wi, P.Wi_b, nullptr, t1, 2048, 1024, 1024, 1024, 0, 0, 0, 0, 1);
        gemm(xcp, K.Wf, P.Wf_b, nullptr, t2, 2048, 1024, 1024, 1024, 0, 0, 0, 0, 1);
        gemm(xlp, K.Wo, P.Wo_b, nullptr, t3, 2048, 1024, 1024, 1024, 0, 0, 0, 0, 1);
        k_mgate<<<128, 256, 0, stream>>>(t1, t2, kb, vb, stC[l], stN[l], stM[l]);
        k_denom<<<1024, 128, 0, stream>>>(stN[l], qb, dnm);
        k_mout<<<256, 128, 0, stream>>>(t3, stC[l], qb, dnm, xs, xr, P.gn_g, P.gn_b, mo);
        packa(mo, mop, 1024, 1024);
        gemm(mop, K.down, P.down_b, X, X, 1024, 1024, 1024, 1024, 0, 0, 0, 0, 1);
      } else {                                 // ---------- sLSTM layer
        const SParams& P = sp[l / 2]; const SPack& K = sk[l / 2];
        k_ln<<<32, 256, 0, stream>>>(X, P.ln_g, P.ln_b, xn);
        k_conv_silu<<<(32768 + 255) / 256, 256, 0, stream>>>(xn, P.conv_w, P.conv_b, xc, 1024, 32768);
        packa(X, Xp, 1024, 1024);
        packa(xc, xcp, 1024, 1024);
        gemm(Xp,    K.Wz, P.Wz_b, nullptr, t1, 128, 128, 128, 1024, 4096, 16384, 128, 128, 8);
        gemm(hp[l], K.Rz, P.Rz_b, t1,      t1, 128, 128, 128, 1024, 4096, 16384, 128, 128, 8);
        gemm(xcp,   K.Wi, P.Wi_b, nullptr, t2, 128, 128, 128, 1024, 4096, 16384, 128, 128, 8);
        gemm(hp[l], K.Ri, P.Ri_b, t2,      t2, 128, 128, 128, 1024, 4096, 16384, 128, 128, 8);
        gemm(xcp,   K.Wf, P.Wf_b, nullptr, t3, 128, 128, 128, 1024, 4096, 16384, 128, 128, 8);
        gemm(hp[l], K.Rf, P.Rf_b, t3,      t3, 128, 128, 128, 1024, 4096, 16384, 128, 128, 8);
        gemm(Xp,    K.Wo, P.Wo_b, nullptr, t4, 128, 128, 128, 1024, 4096, 16384, 128, 128, 8);
        gemm(hp[l], K.Ro, P.Ro_b, t4,      t4, 128, 128, 128, 1024, 4096, 16384, 128, 128, 8);
        k_sgate<<<256, 128, 0, stream>>>(t1, t2, t3, t4, stC[l], stN[l], stM[l], stH[l],
                                         P.gn_g, P.gn_b, gnb);
        packa(stH[l], hp[l], 1024, 1024);
        packa(gnb, gnp, 1024, 1024);
        gemm(gnp, K.upL, P.upL_b, nullptr, uL, 1024, PFS_N, PFS_T, PFS_P, 0, 0, 0, 0, 1);
        gemm(gnp, K.upR, P.upR_b, nullptr, uR, 1024, PFS_N, PFS_T, PFS_P, 0, 0, 0, 0, 1);
        int tot = 32 * PFS_P;
        k_gelu_mul<<<(tot + 255) / 256, 256, 0, stream>>>(uL, uR, vp, tot, PFS_P, PFS_T);
        packa(vp, vpp, PFS_P, PFS_P);
        gemm(vpp, K.down, P.down_b, X, X, PFS_P, 1024, 1024, 1024, 0, 0, 0, 0, 1);
      }
    }
    k_proj<<<1, 256, 0, stream>>>(X, out_w, out_b, (float*)d_out, t);
  }
}
